// NN_Dag_90967407329653
// MI455X (gfx1250) — matrix operand order repaired and checked by reading, not compiled, not run on hardware
//
#include <hip/hip_runtime.h>

// NN-DAG fused per-node MLP for MI455X (gfx1250, wave32, WMMA + async-to-LDS).
// out[b,n] = sigmoid( sigmoid(x@W1_n^T + b1_n) @ Wa_n + ba_n ) @ Wb_n + bb_n
//
//   prep kernels: x -> bf16, W1 -> bf16 (same layout), Wa -> bf16 transposed
//                 [node][m2][m1], all into d_ws.
//   main kernel : per (node, 128-row batch tile) workgroup
//     - global_load_async_to_lds_b128 stages both weight slices into LDS (ASYNCcnt)
//     - GEMM1 (K=128) and GEMM2 (K=256) on v_wmma_f32_16x16x32_bf16 with a
//       2-deep B-fragment pipeline (ds_load pairs stay >=1 full tile ahead)
//     - final K=128 mat-vec fused in C-layout registers + shfl_xor reduction

typedef __attribute__((ext_vector_type(16))) __bf16 v16bf;
typedef __attribute__((ext_vector_type(8)))  __bf16 bf16x8;
typedef __attribute__((ext_vector_type(4)))  __bf16 bf16x4;
typedef __attribute__((ext_vector_type(8)))  float  v8f;

#define D_NODES 128
#define M1      256
#define M2      128
#define BT      128   // batch rows per workgroup (8 waves x 16 rows)

// sigmoid: v_mul + v_exp + v_add + v_rcp (raw-rcp is fine: result is
// immediately rounded to bf16 / multiplied into a bf16-grade dot product).
__device__ __forceinline__ float sigmoidf(float v) {
    return __builtin_amdgcn_rcpf(1.0f + __expf(-v));
}

// Async DMA of 16 bytes per lane: global -> LDS, tracked by ASYNCcnt.
// VDST carries the LDS-relative byte address (low 32 bits of the generic LDS
// pointer), VADDR the 64-bit global address.
__device__ __forceinline__ void async_copy16(unsigned lds_off, const void* gaddr) {
    unsigned long long ga = (unsigned long long)gaddr;
    asm volatile("global_load_async_to_lds_b128 %0, %1, off"
                 :: "v"(lds_off), "v"(ga) : "memory");
}
__device__ __forceinline__ void wait_async0() {
    asm volatile("s_wait_asynccnt 0x0" ::: "memory");
}

// B-fragment (32x16 bf16, rows stored [N][K]): lane l holds column N=l&15,
// element i -> K = kt*32 + 8*(l>>4) + (i%8) + 16*(i/8) (two ds_load_b128).
__device__ __forceinline__ v16bf load_bfrag(const __bf16* base_NK, int Kstride,
                                            int nt, int kt, int ll, int lh) {
    const __bf16* p = base_NK + (nt * 16 + ll) * Kstride + kt * 32 + lh * 8;
    union { bf16x8 h[2]; v16bf v; } u;
    u.h[0] = *(const bf16x8*)p;
    u.h[1] = *(const bf16x8*)(p + 16);
    return u.v;
}

// A-fragment (16x32 bf16) from a row-major [M][K] bf16 buffer:
// lane = M + 16*khalf; element i -> k = 8*lh + (i%8) + 16*(i/8):
// two contiguous 8-element (16B) chunks.
__device__ __forceinline__ v16bf load_afrag_bf16(const __bf16* row, int kt, int lh) {
    const __bf16* p = row + kt * 32 + lh * 8;
    union { bf16x8 h[2]; v16bf v; } u;
    u.h[0] = *(const bf16x8*)p;
    u.h[1] = *(const bf16x8*)(p + 16);
    return u.v;
}

// ---------------- prep kernels (run once per launch into d_ws) ----------------

__global__ __launch_bounds__(256)
void prep_f32_to_bf16(const float* __restrict__ src, __bf16* __restrict__ dst) {
    int i = (blockIdx.x * 256 + threadIdx.x) * 4;   // grid sized to cover exactly
    float4 w = *(const float4*)(src + i);
    bf16x4 v;
    v.x = (__bf16)w.x; v.y = (__bf16)w.y; v.z = (__bf16)w.z; v.w = (__bf16)w.w;
    *(bf16x4*)(dst + i) = v;
}

__global__ __launch_bounds__(256)
void prep_wa_bf16(const float* __restrict__ Wa, __bf16* __restrict__ out) {
    extern __shared__ __align__(16) char psmem[];
    __bf16* t = (__bf16*)psmem;                      // [M2][M1] transposed tile
    const int node = blockIdx.x;
    const float* src = Wa + (size_t)node * (M1 * M2);
    for (int i = threadIdx.x * 4; i < M1 * M2; i += 256 * 4) {
        float4 w = *(const float4*)(src + i);        // coalesced read along m2
        int m1 = i >> 7;
        int m2 = i & 127;
        t[(m2 + 0) * M1 + m1] = (__bf16)w.x;
        t[(m2 + 1) * M1 + m1] = (__bf16)w.y;
        t[(m2 + 2) * M1 + m1] = (__bf16)w.z;
        t[(m2 + 3) * M1 + m1] = (__bf16)w.w;
    }
    __syncthreads();
    __bf16* dst = out + (size_t)node * (M1 * M2);
    for (int i = threadIdx.x * 8; i < M1 * M2; i += 256 * 8) {
        *(uint4*)(dst + i) = *(const uint4*)(t + i); // coalesced 16B writes
    }
}

// ---------------- main fused kernel ----------------

template<bool PREP>
__global__ __launch_bounds__(256)
void nn_dag_main(const float* __restrict__ xf,
                 const __bf16* __restrict__ xb,
                 const float* __restrict__ W1f,
                 const __bf16* __restrict__ W1b,
                 const float* __restrict__ b1,
                 const float* __restrict__ Waf,
                 const __bf16* __restrict__ WaTb,
                 const float* __restrict__ ba,
                 const float* __restrict__ Wb,
                 const float* __restrict__ bb,
                 float* __restrict__ out)
{
    extern __shared__ __align__(32) char smem[];
    __bf16* W1L = (__bf16*)smem;          // [M1=256][K=128]    (B-frag friendly)
    __bf16* WaL = W1L + M1 * 128;         // [M2=128][K=M1=256] (pre-transposed)
    __bf16* H1L = WaL + M2 * M1;          // per-wave [16][256] bf16 staging

    const int tid  = threadIdx.x;
    const int wave = tid >> 5;
    const int lane = tid & 31;
    const int lh   = lane >> 4;
    const int ll   = lane & 15;

    const int node = blockIdx.x >> 5;     // 32 consecutive blocks share a node -> L2 reuse
    const int b0   = (blockIdx.x & 31) * BT;

    if (PREP) {
        // Pure bf16 copy: async DMA straight into LDS, no VGPR round-trip.
        const __bf16* s0 = W1b + (size_t)node * (M1 * 128);
        unsigned l0 = (unsigned)(size_t)W1L;
        for (int i = tid * 8; i < M1 * 128; i += 256 * 8)
            async_copy16(l0 + 2u * i, s0 + i);
        const __bf16* s1 = WaTb + (size_t)node * (M1 * M2);
        unsigned l1 = (unsigned)(size_t)WaL;
        for (int i = tid * 8; i < M1 * M2; i += 256 * 8)
            async_copy16(l1 + 2u * i, s1 + i);
        wait_async0();
    } else {
        // Fallback (small d_ws): convert fp32 weights in-kernel.
        const float* src = W1f + (size_t)node * (M1 * 128);
        for (int i = tid * 4; i < M1 * 128; i += 256 * 4) {
            float4 w = *(const float4*)(src + i);
            bf16x4 v;
            v.x = (__bf16)w.x; v.y = (__bf16)w.y; v.z = (__bf16)w.z; v.w = (__bf16)w.w;
            *(bf16x4*)(W1L + i) = v;
        }
        const float* srca = Waf + (size_t)node * (M1 * M2);
        for (int i = tid * 4; i < M1 * M2; i += 256 * 4) {
            float4 w = *(const float4*)(srca + i);
            int m1 = i >> 7;
            int m2 = i & 127;
            WaL[(m2 + 0) * M1 + m1] = (__bf16)w.x;
            WaL[(m2 + 1) * M1 + m1] = (__bf16)w.y;
            WaL[(m2 + 2) * M1 + m1] = (__bf16)w.z;
            WaL[(m2 + 3) * M1 + m1] = (__bf16)w.w;
        }
    }
    __syncthreads();

    // ---- A-fragments for GEMM1 ----
    const int arow = b0 + wave * 16 + ll;
    v16bf afrag[4];
    if (PREP) {
        const __bf16* xrow = xb + (size_t)arow * 128;
#pragma unroll
        for (int kt = 0; kt < 4; ++kt)
            afrag[kt] = load_afrag_bf16(xrow, kt, lh);
    } else {
        const float* xrow = xf + (size_t)arow * 128;
#pragma unroll
        for (int kt = 0; kt < 4; ++kt) {
            int base = kt * 32 + lh * 8;
            float4 f0 = *(const float4*)(xrow + base);
            float4 f1 = *(const float4*)(xrow + base + 4);
            float4 f2 = *(const float4*)(xrow + base + 16);
            float4 f3 = *(const float4*)(xrow + base + 20);
            v16bf a;
            a[0]  = (__bf16)f0.x; a[1]  = (__bf16)f0.y; a[2]  = (__bf16)f0.z; a[3]  = (__bf16)f0.w;
            a[4]  = (__bf16)f1.x; a[5]  = (__bf16)f1.y; a[6]  = (__bf16)f1.z; a[7]  = (__bf16)f1.w;
            a[8]  = (__bf16)f2.x; a[9]  = (__bf16)f2.y; a[10] = (__bf16)f2.z; a[11] = (__bf16)f2.w;
            a[12] = (__bf16)f3.x; a[13] = (__bf16)f3.y; a[14] = (__bf16)f3.z; a[15] = (__bf16)f3.w;
            afrag[kt] = a;
        }
    }

    __bf16* H1w = H1L + wave * (16 * M1);

    // =========== GEMM1: [16,128] x [128,256] -> 16 C tiles ===========
    // 2-deep B-fragment pipeline: the ds_load pair for tile j+2 is issued
    // before the WMMA of tile j, so waits are dscnt<=2/4 instead of 0.
    v8f acc1[16];
#pragma unroll
    for (int i = 0; i < 16; ++i) acc1[i] = (v8f){};

    {
        v16bf bq0 = load_bfrag(W1L, 128, 0, 0, ll, lh);
        v16bf bq1 = load_bfrag(W1L, 128, 1, 0, ll, lh);
#pragma unroll
        for (int j = 0; j < 64; ++j) {
            const int kt = j >> 4;
            const int nt = j & 15;
            const int jp = (j + 2) & 63;
            v16bf bpre = load_bfrag(W1L, 128, jp & 15, jp >> 4, ll, lh);
            acc1[nt] = __builtin_amdgcn_wmma_f32_16x16x32_bf16(
                           false, afrag[kt], false, bq0, (short)0, acc1[nt], false, false);
            bq0 = bq1;
            bq1 = bpre;
        }
    }

    // Bias + sigmoid in C layout (VGPR r, lane -> M=r+8*lh, N=nt*16+ll), store bf16.
#pragma unroll
    for (int nt = 0; nt < 16; ++nt) {
        float bias = b1[node * M1 + nt * 16 + ll];
#pragma unroll
        for (int r = 0; r < 8; ++r) {
            H1w[(r + 8 * lh) * M1 + nt * 16 + ll] = (__bf16)sigmoidf(acc1[nt][r] + bias);
        }
    }
    __syncthreads();

    // =========== GEMM2: [16,256] x [256,128] -> 8 C tiles, fused final dot ===========
    v16bf a2frag[8];
#pragma unroll
    for (int kt = 0; kt < 8; ++kt)
        a2frag[kt] = load_afrag_bf16(H1w + ll * M1, kt, lh);

    v8f acc2[8];
#pragma unroll
    for (int i = 0; i < 8; ++i) acc2[i] = (v8f){};

    {
        v16bf bq0 = load_bfrag(WaL, M1, 0, 0, ll, lh);
        v16bf bq1 = load_bfrag(WaL, M1, 1, 0, ll, lh);
#pragma unroll
        for (int j = 0; j < 64; ++j) {
            const int kt = j >> 3;
            const int nt = j & 7;
            const int jp = (j + 2) & 63;
            v16bf bpre = load_bfrag(WaL, M1, jp & 7, jp >> 3, ll, lh);
            acc2[nt] = __builtin_amdgcn_wmma_f32_16x16x32_bf16(
                           false, a2frag[kt], false, bq0, (short)0, acc2[nt], false, false);
            bq0 = bq1;
            bq1 = bpre;
        }
    }

    float dot[8];
#pragma unroll
    for (int r = 0; r < 8; ++r) dot[r] = 0.0f;

#pragma unroll
    for (int nt = 0; nt < 8; ++nt) {
        float bav = ba[node * M2 + nt * 16 + ll];
        float wbv = Wb[node * M2 + nt * 16 + ll];
#pragma unroll
        for (int r = 0; r < 8; ++r) {
            dot[r] += sigmoidf(acc2[nt][r] + bav) * wbv;
        }
    }

    // Reduce over the 16 lanes of each half-wave (masks <16 stay within halves).
#pragma unroll
    for (int r = 0; r < 8; ++r) {
        float v = dot[r];
        v += __shfl_xor(v, 1, 32);
        v += __shfl_xor(v, 2, 32);
        v += __shfl_xor(v, 4, 32);
        v += __shfl_xor(v, 8, 32);
        dot[r] = v;
    }

    if (ll == 0) {   // lanes 0 / 16 hold row groups M=0..7 / 8..15
        float bbv = bb[node];
        int rbase = b0 + wave * 16 + 8 * lh;
#pragma unroll
        for (int r = 0; r < 8; ++r) {
            out[(size_t)(rbase + r) * D_NODES + node] = dot[r] + bbv;
        }
    }
}

// ---------------- host launcher ----------------

extern "C" void kernel_launch(void* const* d_in, const int* in_sizes, int n_in,
                              void* d_out, int out_size, void* d_ws, size_t ws_size,
                              hipStream_t stream) {
    (void)in_sizes; (void)n_in; (void)out_size;

    const float* x  = (const float*)d_in[0];
    const float* W1 = (const float*)d_in[1];
    const float* b1 = (const float*)d_in[2];
    const float* Wa = (const float*)d_in[3];
    const float* ba = (const float*)d_in[4];
    const float* Wb = (const float*)d_in[5];
    const float* bb = (const float*)d_in[6];
    float* out = (float*)d_out;

    const size_t n_w1 = (size_t)D_NODES * M1 * 128;   // 4,194,304
    const size_t n_wa = (size_t)D_NODES * M1 * M2;    // 4,194,304
    const size_t n_x  = (size_t)4096 * 128;           //   524,288
    const size_t ws_needed = (n_w1 + n_wa + n_x) * sizeof(__bf16);   // ~17 MB
    const bool use_prep = (ws_size >= ws_needed) && (d_ws != nullptr);

    // Dynamic LDS: W1 slice (64 KB) + Wa slice (64 KB) + 8x per-wave H1 (64 KB).
    const int lds_bytes = (int)((M1 * 128 + M2 * M1 + 8 * 16 * M1) * sizeof(__bf16));

    dim3 grid(D_NODES * (4096 / BT));   // 128 nodes * 32 batch tiles
    dim3 block(256);

    if (use_prep) {
        __bf16* W1b  = (__bf16*)d_ws;
        __bf16* WaTb = W1b + n_w1;
        __bf16* xb   = WaTb + n_wa;
        prep_f32_to_bf16<<<dim3((unsigned)(n_w1 / (256 * 4))), block, 0, stream>>>(W1, W1b);
        prep_f32_to_bf16<<<dim3((unsigned)(n_x  / (256 * 4))), block, 0, stream>>>(x, xb);
        prep_wa_bf16<<<dim3(D_NODES), block, M1 * M2 * sizeof(__bf16), stream>>>(Wa, WaTb);

        hipFuncSetAttribute((const void*)&nn_dag_main<true>,
                            hipFuncAttributeMaxDynamicSharedMemorySize, lds_bytes);
        nn_dag_main<true><<<grid, block, lds_bytes, stream>>>(
            x, xb, W1, W1b, b1, Wa, WaTb, ba, Wb, bb, out);
    } else {
        hipFuncSetAttribute((const void*)&nn_dag_main<false>,
                            hipFuncAttributeMaxDynamicSharedMemorySize, lds_bytes);
        nn_dag_main<false><<<grid, block, lds_bytes, stream>>>(
            x, nullptr, W1, nullptr, b1, Wa, nullptr, ba, Wb, bb, out);
    }
}
